// OperationFeatureBlock_71425306133049
// MI455X (gfx1250) — matrix operand order repaired and checked by reading, not compiled, not run on hardware
//
#include <hip/hip_runtime.h>
#include <hip/hip_bf16.h>

#define Bsz 8
#define Cn  2048
#define Dn  128
#define Jn  128
#define CM2 2046   // C-2

typedef __attribute__((ext_vector_type(16))) __bf16 v16bf;
typedef __attribute__((ext_vector_type(8)))  __bf16 v8bf;
typedef __attribute__((ext_vector_type(8)))  float  v8f;

// ---- WMMA helpers -----------------------------------------------------------
// A-matrix (16x32 bf16) k order chosen here (B uses the same): lane<16 -> row=lane, K 0..15 contiguous;
// lane>=16 -> row=lane-16, K 16..31.  We fold the +16 element offset
// into the per-lane base pointer, so each lane does two 16B loads: [0..7],[8..15].
static __device__ __forceinline__ v16bf cat8(v8bf lo, v8bf hi) {
  v16bf r;
#pragma unroll
  for (int i = 0; i < 8; ++i) { r[i] = lo[i]; r[i + 8] = hi[i]; }
  return r;
}
static __device__ __forceinline__ v16bf ldA(const __bf16* p) {
  return cat8(*(const v8bf*)p, *(const v8bf*)(p + 8));
}
// B-matrix (32x16 bf16) k order chosen here (matches A): lane<16 -> col=lane, K 0..15 contiguous;
// lane>=16 -> col=lane-16, K 16..31.  +16-element offset folded into base ptr.
static __device__ __forceinline__ v16bf ldB16(const __bf16* p) {
  return cat8(*(const v8bf*)p, *(const v8bf*)(p + 8));
}
static __device__ __forceinline__ v8f wmma_bf16(v16bf a, v16bf b, v8f c) {
  return __builtin_amdgcn_wmma_f32_16x16x32_bf16(false, a, false, b, (short)0, c,
                                                 false, false);
}

// ---- prep kernels -----------------------------------------------------------
__global__ void conv_ops_kernel(const float* __restrict__ ops,
                                __bf16* __restrict__ ops_bf, int n) {
  int i = blockIdx.x * blockDim.x + threadIdx.x;
  if (i < n) ops_bf[i] = (__bf16)ops[i];
}

// Transposed bf16 weights so B-matrix loads are K-contiguous per output column.
__global__ void prep_w_kernel(const float* __restrict__ W_qkv,
                              const float* __restrict__ W_out,
                              const float* __restrict__ W_mix2,
                              const float* __restrict__ W_mix3,
                              __bf16* __restrict__ WvT, __bf16* __restrict__ WoutT,
                              __bf16* __restrict__ Wmix2T, __bf16* __restrict__ Wmix3T) {
  int i = blockIdx.x * blockDim.x + threadIdx.x;
  if (i < 128 * 128) {
    int n = i >> 7, k = i & 127;
    WvT[n * 128 + k]   = (__bf16)W_qkv[k * 384 + 256 + n];  // V part of W_qkv
    WoutT[n * 128 + k] = (__bf16)W_out[k * 128 + n];
  }
  if (i < 128 * 256) {
    int n = i >> 8, kk = i & 255;
    Wmix2T[n * 256 + kk] = (__bf16)W_mix2[kk * 128 + n];
  }
  if (i < 128 * 384) {
    int n = i / 384, kk = i % 384;
    Wmix3T[n * 384 + kk] = (__bf16)W_mix3[kk * 128 + n];
  }
}

// Rank-1 score decomposition: wq_eff = W_q w_q, wk_eff = W_k w_k,
// sbias = b_score + b_q.w_q + b_k.w_k
__global__ void prep_eff_kernel(const float* __restrict__ W_qkv,
                                const float* __restrict__ b_qkv,
                                const float* __restrict__ w_score,
                                const float* __restrict__ b_score,
                                float* __restrict__ wq_eff, float* __restrict__ wk_eff,
                                float* __restrict__ sbias) {
  int t = threadIdx.x;
  if (t < 128) {
    float aq = 0.f, ak = 0.f;
    for (int e = 0; e < 128; ++e) {
      aq += W_qkv[t * 384 + e] * w_score[e];
      ak += W_qkv[t * 384 + 128 + e] * w_score[128 + e];
    }
    wq_eff[t] = aq; wk_eff[t] = ak;
  } else if (t == 128) {
    float s = b_score[0];
    for (int e = 0; e < 128; ++e)
      s += b_qkv[e] * w_score[e] + b_qkv[128 + e] * w_score[128 + e];
    sbias[0] = s;
  }
}

// sq/sk: one wave per row (matrix-vector).
__global__ void sqsk_kernel(const float* __restrict__ ops,
                            const float* __restrict__ wq_eff,
                            const float* __restrict__ wk_eff,
                            float* __restrict__ sq, float* __restrict__ sk) {
  int row = blockIdx.x * 8 + (threadIdx.x >> 5);
  int lane = threadIdx.x & 31;
  const float* r = ops + (size_t)row * Dn;
  float a = 0.f, c = 0.f;
  for (int k = lane; k < Dn; k += 32) { float x = r[k]; a += x * wq_eff[k]; c += x * wk_eff[k]; }
  for (int off = 16; off > 0; off >>= 1) { a += __shfl_down(a, off, 32); c += __shfl_down(c, off, 32); }
  if (lane == 0) { sq[row] = a; sk[row] = c; }
}

// ---- V projection (WMMA), stored transposed bf16: vT[b][d][c] --------------
__global__ __launch_bounds__(128) void v_kernel(const __bf16* __restrict__ ops_bf,
                                                const __bf16* __restrict__ WvT,
                                                const float* __restrict__ b_qkv,
                                                __bf16* __restrict__ vT) {
  int blk = blockIdx.x;
  int b = blk >> 7, i0 = (blk & 127) << 4;
  int t = threadIdx.x, lane = t & 31, wave = t >> 5;
  int hl = lane & 15, hi = lane >> 4, n0 = wave * 32;
  const __bf16* Ab = ops_bf + ((size_t)b * Cn + i0 + hl) * Dn + hi * 16;
  const __bf16* B0 = WvT + (size_t)(n0 + hl) * Dn + hi * 16;
  const __bf16* B1 = WvT + (size_t)(n0 + 16 + hl) * Dn + hi * 16;
  v8f a0 = {}, a1 = {};
#pragma unroll
  for (int k0 = 0; k0 < 128; k0 += 32) {
    v16bf A = ldA(Ab + k0);
    a0 = wmma_bf16(A, ldB16(B0 + k0), a0);
    a1 = wmma_bf16(A, ldB16(B1 + k0), a1);
  }
  float c0 = b_qkv[256 + n0 + hl], c1 = b_qkv[256 + n0 + 16 + hl];
  __bf16* vb = vT + (size_t)b * Dn * Cn;
#pragma unroll
  for (int r = 0; r < 8; ++r) {
    int m = i0 + r + hi * 8;
    vb[(size_t)(n0 + hl) * Cn + m]      = (__bf16)(a0[r] + c0);
    vb[(size_t)(n0 + 16 + hl) * Cn + m] = (__bf16)(a1[r] + c1);
  }
}

// ---- gathered mix3 GEMM (WMMA): seq[:, :C-2] --------------------------------
__global__ __launch_bounds__(128) void mix3_kernel(const __bf16* __restrict__ ops_bf,
                                                   const int* __restrict__ relations,
                                                   const __bf16* __restrict__ Wmix3T,
                                                   const float* __restrict__ b_mix3,
                                                   __bf16* __restrict__ seq_bf) {
  int blk = blockIdx.x;
  int b = blk >> 7, i0 = (blk & 127) << 4;
  int t = threadIdx.x, lane = t & 31, wave = t >> 5;
  int hl = lane & 15, hi = lane >> 4, n0 = wave * 32;
  int row = i0 + hl;
  int rr = row < CM2 ? row : CM2 - 1;
  int p0 = relations[((size_t)b * CM2 + rr) * 2 + 0];
  int p1 = relations[((size_t)b * CM2 + rr) * 2 + 1];
  const __bf16* base = ops_bf + (size_t)b * Cn * Dn;
  const __bf16* Aseg[3] = { base + (size_t)row * Dn + hi * 16,
                            base + (size_t)p0 * Dn + hi * 16,
                            base + (size_t)p1 * Dn + hi * 16 };
  const __bf16* Bb0 = Wmix3T + (size_t)(n0 + hl) * 384 + hi * 16;
  const __bf16* Bb1 = Wmix3T + (size_t)(n0 + 16 + hl) * 384 + hi * 16;
  v8f a0 = {}, a1 = {};
#pragma unroll
  for (int seg = 0; seg < 3; ++seg) {
    const __bf16* Ab = Aseg[seg];
#pragma unroll
    for (int k0 = 0; k0 < 128; k0 += 32) {
      v16bf A = ldA(Ab + k0);
      a0 = wmma_bf16(A, ldB16(Bb0 + seg * 128 + k0), a0);
      a1 = wmma_bf16(A, ldB16(Bb1 + seg * 128 + k0), a1);
    }
  }
  float c0 = b_mix3[n0 + hl], c1 = b_mix3[n0 + 16 + hl];
#pragma unroll
  for (int r = 0; r < 8; ++r) {
    int m = i0 + r + hi * 8;
    if (m < CM2) {
      seq_bf[((size_t)b * Cn + m) * Dn + n0 + hl]      = (__bf16)(a0[r] + c0);
      seq_bf[((size_t)b * Cn + m) * Dn + n0 + 16 + hl] = (__bf16)(a1[r] + c1);
    }
  }
}

// ---- begin/end rows: (mean of gathered rows) @ W + b ------------------------
__global__ __launch_bounds__(128) void beginend_kernel(const float* __restrict__ ops,
                                                       const int* __restrict__ begins,
                                                       const int* __restrict__ ends,
                                                       const float* __restrict__ W_begin,
                                                       const float* __restrict__ b_begin,
                                                       const float* __restrict__ W_end,
                                                       const float* __restrict__ b_end,
                                                       __bf16* __restrict__ seq_bf) {
  int b = blockIdx.x >> 1, which = blockIdx.x & 1;
  const int* idx = which ? ends + b * Jn : begins + b * Jn;
  const float* W  = which ? W_end : W_begin;
  const float* bb = which ? b_end : b_begin;
  __shared__ float m[Dn];
  int t = threadIdx.x;
  float acc = 0.f;
  for (int j = 0; j < Jn; ++j) acc += ops[((size_t)b * Cn + idx[j]) * Dn + t];
  m[t] = acc * (1.f / (float)Jn);
  __syncthreads();
  float o = bb[t];
  for (int d = 0; d < Dn; ++d) o += m[d] * W[d * Dn + t];
  seq_bf[((size_t)b * Cn + (CM2 + which)) * Dn + t] = (__bf16)o;
}

// ---- fused attention + W_out + mix2 (single pass over the 134MB mask) -------
#define ATTN_SMEM (16 * Cn * 2 + 128 * 4 + 16 * 4 + 16 * Dn * 2 + 16 * Dn * 2)

__global__ __launch_bounds__(128) void attn_kernel(const float* __restrict__ sq,
                                                   const float* __restrict__ sk,
                                                   const float* __restrict__ sbias,
                                                   const float* __restrict__ mask,
                                                   const __bf16* __restrict__ vT,
                                                   const __bf16* __restrict__ seq_bf,
                                                   const __bf16* __restrict__ WoutT,
                                                   const __bf16* __restrict__ Wmix2T,
                                                   const float* __restrict__ b_out,
                                                   const float* __restrict__ b_mix2,
                                                   float* __restrict__ out) {
  extern __shared__ char smem[];
  __bf16* P        = (__bf16*)smem;                      // [16][2048] bf16, 64KB
  float*  red      = (float*)(smem + 16 * Cn * 2);       // [128]
  float*  rowScale = red + 128;                          // [16]
  __bf16* T1       = (__bf16*)(rowScale + 16);           // [16][128] weighted
  __bf16* T2       = T1 + 16 * Dn;                       // [16][128] same_type

  int b = blockIdx.x >> 7;
  int i0 = (blockIdx.x & 127) << 4;
  int t = threadIdx.x, lane = t & 31, wave = t >> 5;
  int hl = lane & 15, hi = lane >> 4, n0 = wave * 32;

  float bias = sbias[0];
  // sk stripe is identical across the 16 rows -> preload once per thread
  float skv[16];
#pragma unroll
  for (int jj = 0; jj < 16; ++jj) skv[jj] = sk[(size_t)b * Cn + jj * 128 + t];

  const float* maskB = mask + ((size_t)b * Cn + i0) * Cn;

  // -------- phase 1: masked leaky-relu scores -> softmax probs (bf16 in LDS)
  for (int i = 0; i < 16; ++i) {
    float sqi = sq[(size_t)b * Cn + i0 + i] + bias;
    const float* mrow = maskB + (size_t)i * Cn;
    float sv[16];
    float mx = -1e30f;
#pragma unroll
    for (int jj = 0; jj < 16; ++jj) {
      float s = sqi + skv[jj];
      s = (s > 0.f) ? s : 0.01f * s;          // leaky_relu (slope 0.01)
      s *= mrow[jj * 128 + t];                // multiplicative mask, pre-softmax
      sv[jj] = s;
      mx = fmaxf(mx, s);
    }
    red[t] = mx; __syncthreads();
    for (int off = 64; off > 0; off >>= 1) {
      if (t < off) red[t] = fmaxf(red[t], red[t + off]);
      __syncthreads();
    }
    mx = red[0];
    float sum = 0.f;
#pragma unroll
    for (int jj = 0; jj < 16; ++jj) {
      float p = __expf(sv[jj] - mx);
      sum += p;
      P[i * Cn + jj * 128 + t] = (__bf16)p;
    }
    __syncthreads();
    red[t] = sum; __syncthreads();
    for (int off = 64; off > 0; off >>= 1) {
      if (t < off) red[t] += red[t + off];
      __syncthreads();
    }
    if (t == 0) rowScale[i] = 1.f / red[0];
    __syncthreads();
  }

  // -------- phase 2: weighted = softmax(P) @ V   (K = 2048, bf16 WMMA)
  const __bf16* Arow = P + (size_t)hl * Cn + hi * 16;
  const __bf16* vb   = vT + (size_t)b * Dn * Cn;
  const __bf16* Bv0  = vb + (size_t)(n0 + hl) * Cn + hi * 16;
  const __bf16* Bv1  = vb + (size_t)(n0 + 16 + hl) * Cn + hi * 16;
  v8f acc0 = {}, acc1 = {};
  for (int k0 = 0; k0 < Cn; k0 += 32) {
    v16bf A = ldA(Arow + k0);
    acc0 = wmma_bf16(A, ldB16(Bv0 + k0), acc0);
    acc1 = wmma_bf16(A, ldB16(Bv1 + k0), acc1);
  }
  // softmax divide folded into accumulator scaling; stage weighted tile in LDS
#pragma unroll
  for (int r = 0; r < 8; ++r) {
    int m = r + hi * 8;
    float sc = rowScale[m];
    T1[m * Dn + n0 + hl]      = (__bf16)(acc0[r] * sc);
    T1[m * Dn + n0 + 16 + hl] = (__bf16)(acc1[r] * sc);
  }
  __syncthreads();

  // -------- phase 3: same_type = weighted @ W_out + b_out  (K = 128)
  const __bf16* A2  = T1 + hl * Dn + hi * 16;
  const __bf16* Bo0 = WoutT + (size_t)(n0 + hl) * Dn + hi * 16;
  const __bf16* Bo1 = WoutT + (size_t)(n0 + 16 + hl) * Dn + hi * 16;
  v8f a2_0 = {}, a2_1 = {};
#pragma unroll
  for (int k0 = 0; k0 < 128; k0 += 32) {
    v16bf A = ldA(A2 + k0);
    a2_0 = wmma_bf16(A, ldB16(Bo0 + k0), a2_0);
    a2_1 = wmma_bf16(A, ldB16(Bo1 + k0), a2_1);
  }
  float bo0 = b_out[n0 + hl], bo1 = b_out[n0 + 16 + hl];
#pragma unroll
  for (int r = 0; r < 8; ++r) {
    int m = r + hi * 8;
    T2[m * Dn + n0 + hl]      = (__bf16)(a2_0[r] + bo0);
    T2[m * Dn + n0 + 16 + hl] = (__bf16)(a2_1[r] + bo1);
  }
  __syncthreads();

  // -------- phase 4: out = seq @ W_mix2[:D] + same_type @ W_mix2[D:] + b_mix2
  v8f a3_0 = {}, a3_1 = {};
  const __bf16* Aseq = seq_bf + ((size_t)b * Cn + i0 + hl) * Dn + hi * 16;
  const __bf16* Bm0  = Wmix2T + (size_t)(n0 + hl) * 256 + hi * 16;
  const __bf16* Bm1  = Wmix2T + (size_t)(n0 + 16 + hl) * 256 + hi * 16;
#pragma unroll
  for (int k0 = 0; k0 < 128; k0 += 32) {
    v16bf A = ldA(Aseq + k0);
    a3_0 = wmma_bf16(A, ldB16(Bm0 + k0), a3_0);
    a3_1 = wmma_bf16(A, ldB16(Bm1 + k0), a3_1);
  }
  const __bf16* A3 = T2 + hl * Dn + hi * 16;
#pragma unroll
  for (int k0 = 0; k0 < 128; k0 += 32) {
    v16bf A = ldA(A3 + k0);
    a3_0 = wmma_bf16(A, ldB16(Bm0 + 128 + k0), a3_0);
    a3_1 = wmma_bf16(A, ldB16(Bm1 + 128 + k0), a3_1);
  }
  float bm0 = b_mix2[n0 + hl], bm1 = b_mix2[n0 + 16 + hl];
#pragma unroll
  for (int r = 0; r < 8; ++r) {
    int m = r + hi * 8;
    out[((size_t)b * Cn + i0 + m) * Dn + n0 + hl]      = a3_0[r] + bm0;
    out[((size_t)b * Cn + i0 + m) * Dn + n0 + 16 + hl] = a3_1[r] + bm1;
  }
}

// ---- launch -----------------------------------------------------------------
extern "C" void kernel_launch(void* const* d_in, const int* in_sizes, int n_in,
                              void* d_out, int out_size, void* d_ws, size_t ws_size,
                              hipStream_t stream) {
  (void)in_sizes; (void)n_in; (void)out_size; (void)ws_size;
  const float* ops      = (const float*)d_in[0];
  const int*   relations= (const int*)d_in[1];
  const int*   begins   = (const int*)d_in[2];
  const int*   ends     = (const int*)d_in[3];
  const float* mask     = (const float*)d_in[4];
  const float* W_begin  = (const float*)d_in[5];
  const float* b_begin  = (const float*)d_in[6];
  const float* W_end    = (const float*)d_in[7];
  const float* b_end    = (const float*)d_in[8];
  const float* W_mix3   = (const float*)d_in[9];
  const float* b_mix3   = (const float*)d_in[10];
  const float* W_qkv    = (const float*)d_in[11];
  const float* b_qkv    = (const float*)d_in[12];
  const float* w_score  = (const float*)d_in[13];
  const float* b_score  = (const float*)d_in[14];
  const float* W_out    = (const float*)d_in[15];
  const float* b_out    = (const float*)d_in[16];
  const float* W_mix2   = (const float*)d_in[17];
  const float* b_mix2   = (const float*)d_in[18];
  float* out = (float*)d_out;

  char* ws = (char*)d_ws;
  __bf16* ops_bf = (__bf16*)(ws + 0);                 // 4,194,304 B
  __bf16* vT     = (__bf16*)(ws + 4194304);           // 4,194,304 B
  __bf16* seq_bf = (__bf16*)(ws + 8388608);           // 4,194,304 B
  float*  sq     = (float*) (ws + 12582912);          //    65,536 B
  float*  sk     = (float*) (ws + 12648448);          //    65,536 B
  __bf16* WvT    = (__bf16*)(ws + 12713984);          //    32,768 B
  __bf16* WoutT  = (__bf16*)(ws + 12746752);          //    32,768 B
  __bf16* Wmix2T = (__bf16*)(ws + 12779520);          //    65,536 B
  __bf16* Wmix3T = (__bf16*)(ws + 12845056);          //    98,304 B
  float*  wq_eff = (float*) (ws + 12943360);
  float*  wk_eff = (float*) (ws + 12943872);
  float*  sbias  = (float*) (ws + 12944384);

  conv_ops_kernel<<<(Bsz * Cn * Dn + 255) / 256, 256, 0, stream>>>(ops, ops_bf, Bsz * Cn * Dn);
  prep_w_kernel<<<192, 256, 0, stream>>>(W_qkv, W_out, W_mix2, W_mix3, WvT, WoutT, Wmix2T, Wmix3T);
  prep_eff_kernel<<<1, 256, 0, stream>>>(W_qkv, b_qkv, w_score, b_score, wq_eff, wk_eff, sbias);
  sqsk_kernel<<<(Bsz * Cn) / 8, 256, 0, stream>>>(ops, wq_eff, wk_eff, sq, sk);
  v_kernel<<<Bsz * (Cn / 16), 128, 0, stream>>>(ops_bf, WvT, b_qkv, vT);
  mix3_kernel<<<Bsz * (Cn / 16), 128, 0, stream>>>(ops_bf, relations, Wmix3T, b_mix3, seq_bf);
  beginend_kernel<<<Bsz * 2, 128, 0, stream>>>(ops, begins, ends, W_begin, b_begin, W_end, b_end, seq_bf);
  attn_kernel<<<Bsz * (Cn / 16), 128, ATTN_SMEM, stream>>>(sq, sk, sbias, mask, vT, seq_bf,
                                                           WoutT, Wmix2T, b_out, b_mix2, out);
}
